// DPRNNBlock_2989297238423
// MI455X (gfx1250) — compile-verified, matrix operand order checked by reading, not run on hardware
//
#include <hip/hip_runtime.h>
#include <hip/hip_bf16.h>
#include <stdint.h>

// ---------------- problem constants ----------------
#define Bc   4
#define Cc   64
#define Tc   128
#define Fc   128
#define Hc   128          // hidden per direction
#define Kk   8            // unfold / deconv kernel
#define Lc   (Fc - Kk + 1)   // 121
#define Nn   (Bc * Tc)       // 512
#define Mrows ((size_t)Lc * Nn)  // 61952  (divisible by 16: 3872 tiles)
#define FOURH 512         // 4*H
#define OUTW  1024        // 2 dirs * 4H  (GEMM output width)
#define KC   64           // GEMM K-chunk staged in LDS (double buffered)
#define BPAD (KC + 8)     // padded LDS row stride: 36 dwords -> conflict-free

typedef __attribute__((ext_vector_type(16))) _Float16 v16h;
typedef __attribute__((ext_vector_type(8)))  _Float16 v8h;
typedef __attribute__((ext_vector_type(8)))  float    v8f;

// async copy of 16 bytes global->LDS (CDNA5, ASYNCcnt-tracked).
// offset: applies to BOTH lds and global address (contiguous copy).
#define ASYNC_CP16(ldsaddr, gaddr, off)                                        \
    asm volatile("global_load_async_to_lds_b128 %0, %1, off offset:" #off      \
                 :: "v"(ldsaddr), "v"(gaddr) : "memory")

// pin a loaded value in registers at this point (prevents sinking past barriers)
#define PIN(x) asm volatile("" : "+v"(x))

#define SHUF16(a, b) __builtin_shufflevector(a, b, 0,1,2,3,4,5,6,7,8,9,10,11,12,13,14,15)

// ---------------- 1. channel norm over C ----------------
__global__ __launch_bounds__(256)
void chan_norm(const float* __restrict__ x, const float* __restrict__ gamma,
               const float* __restrict__ beta, float* __restrict__ xn) {
    int idx = blockIdx.x * 256 + threadIdx.x;        // over B*T*F = 65536
    int f = idx & (Fc - 1);
    int t = (idx >> 7) & (Tc - 1);
    int b = idx >> 14;
    const size_t cs = (size_t)Tc * Fc;               // stride between channels
    const float* xp = x + (size_t)b * Cc * cs + (size_t)t * Fc + f;
    float s = 0.f, s2 = 0.f;
    for (int c = 0; c < Cc; ++c) {
        float v = xp[c * cs];
        s += v; s2 += v * v;
    }
    float mu  = s * (1.f / Cc);
    float var = s2 * (1.f / Cc) - mu * mu;
    float inv = rsqrtf(var + 1e-8f);
    float* op = xn + (size_t)b * Cc * cs + (size_t)t * Fc + f;
    for (int c = 0; c < Cc; ++c) {
        float v = xp[c * cs];
        op[c * cs] = gamma[c] * (v - mu) * inv + beta[c];
    }
}

// ---------------- 2. unfold -> h0 (L, N, C*K) f16 ----------------
__global__ __launch_bounds__(256)
void unfold_k(const float* __restrict__ xn, _Float16* __restrict__ h0) {
    size_t idx = (size_t)blockIdx.x * 256 + threadIdx.x;   // over L*N*512
    int d = (int)(idx & 511);            // c*K + k
    int n = (int)((idx >> 9) & (Nn - 1));
    int l = (int)(idx >> 18);            // N*512 = 2^18
    int c = d >> 3, k = d & 7;
    int b = n >> 7, t = n & (Tc - 1);
    float v = xn[(((size_t)b * Cc + c) * Tc + t) * Fc + (l + k)];
    h0[idx] = (_Float16)v;
}

// ---------------- 3. weight convert+transpose: (Kd,2,4H) f32 -> (1024,Kd) f16 ----
__global__ __launch_bounds__(256)
void wconvert(const float* __restrict__ W, _Float16* __restrict__ Wt, int Kd) {
    int idx = blockIdx.x * 256 + threadIdx.x;   // over 1024*Kd
    if (idx >= OUTW * Kd) return;
    int i = idx % Kd;          // input-dim (K of gemm)
    int j = idx / Kd;          // output col (dir*512 + 4H idx)
    Wt[idx] = (_Float16)W[(size_t)i * OUTW + j];
}

// ---------------- 4. WMMA GEMM: U = A(M x Kd) * Wt^T -> (M x 1024) f16 ----------
// Block tile 128(M) x 128(N): 8 waves, each wave owns 2 M-tiles x 4 N-tiles
// (wv&3 -> M-group, wv>>2 -> N-half). Weight strip staged into LDS in KC
// chunks, DOUBLE-BUFFERED with async-to-LDS; s_wait_asynccnt 4 retires exactly
// the older chunk (async loads complete in order). The chunk's A fragments are
// loaded and pinned BEFORE the async wait/barrier so global latency overlaps
// the pipeline. Each LDS B fragment feeds two back-to-back WMMAs.
__global__ __launch_bounds__(256)
void sru_gemm(const _Float16* __restrict__ A, const _Float16* __restrict__ Bt,
              _Float16* __restrict__ U, int Kd) {
    __shared__ __align__(16) _Float16 sB[2][128 * BPAD];   // 2 x 18,432 B
    const int tid    = threadIdx.x;
    const int lane   = tid & 31;
    const int wv     = tid >> 5;                     // 0..7
    const int mTile0 = blockIdx.x * 8 + (wv & 3) * 2;   // 2 consecutive M-tiles
    const int ncol0  = blockIdx.y * 128 + (wv >> 2) * 64;  // 4 N-tiles
    const int r16    = lane & 15;
    const int kb     = (lane >> 4) * 8;

    // staging role: thread t copies row (t>>1), half-range [(t&1)*32, +32)
    const int srow  = tid >> 1;
    const int spart = (tid & 1) * 32;

    v8f acc[2][4] = {};
    const size_t arow0 = ((size_t)mTile0 * 16 + r16) * Kd;   // A row, tile 0
    const size_t arow1 = arow0 + (size_t)16 * Kd;            // A row, tile 1
    const _Float16* sbase = Bt + (size_t)(blockIdx.y * 128 + srow) * Kd + spart;
    const uint32_t dst0 = (uint32_t)(uintptr_t)&sB[0][srow * BPAD + spart];
    const uint32_t dst1 = (uint32_t)(uintptr_t)&sB[1][srow * BPAD + spart];
    const int bcol = (wv >> 2) * 64 + r16;   // LDS column base for this wave

    // prologue: stage chunk 0 into buffer 0 (4 async b128 per thread)
    {
        uint64_t ga = (uint64_t)(uintptr_t)sbase;
        ASYNC_CP16(dst0, ga, 0);  ASYNC_CP16(dst0, ga, 16);
        ASYNC_CP16(dst0, ga, 32); ASYNC_CP16(dst0, ga, 48);
    }

    const int nch = Kd / KC;
    for (int i = 0; i < nch; ++i) {
        const int kc = i * KC;
        const _Float16* scur = (i & 1) ? &sB[1][0] : &sB[0][0];
        if (i + 1 < nch) {
            const uint32_t dnxt = (i & 1) ? dst0 : dst1;
            uint64_t ga = (uint64_t)(uintptr_t)(sbase + kc + KC);
            ASYNC_CP16(dnxt, ga, 0);  ASYNC_CP16(dnxt, ga, 16);
            ASYNC_CP16(dnxt, ga, 32); ASYNC_CP16(dnxt, ga, 48);
        }

        // ---- A fragments for BOTH K-steps of this chunk, pinned above wait ----
        const _Float16* a0p = A + arow0 + kc;
        const _Float16* a1p = A + arow1 + kc;
        v8h a00 = *(const v8h*)(a0p + kb);        v8h a01 = *(const v8h*)(a0p + kb + 16);
        v8h a10 = *(const v8h*)(a1p + kb);        v8h a11 = *(const v8h*)(a1p + kb + 16);
        v8h a02 = *(const v8h*)(a0p + 32 + kb);   v8h a03 = *(const v8h*)(a0p + 32 + kb + 16);
        v8h a12 = *(const v8h*)(a1p + 32 + kb);   v8h a13 = *(const v8h*)(a1p + 32 + kb + 16);
        PIN(a00); PIN(a01); PIN(a10); PIN(a11);
        PIN(a02); PIN(a03); PIN(a12); PIN(a13);

        if (i + 1 < nch) {
            asm volatile("s_wait_asynccnt 0x4" ::: "memory");  // chunk i landed
        } else {
            asm volatile("s_wait_asynccnt 0x0" ::: "memory");
        }
        __syncthreads();

        // ---- K-step 0 ----
        {
            v16h af0 = SHUF16(a00, a01);
            v16h af1 = SHUF16(a10, a11);
#pragma unroll
            for (int nt = 0; nt < 4; ++nt) {
                const _Float16* bp = scur + (nt * 16 + bcol) * BPAD + kb;
                v8h b0 = *(const v8h*)(bp);
                v8h b1 = *(const v8h*)(bp + 16);
                v16h bfr = SHUF16(b0, b1);
                acc[0][nt] = __builtin_amdgcn_wmma_f32_16x16x32_f16(
                    false, af0, false, bfr, (short)0, acc[0][nt], false, false);
                acc[1][nt] = __builtin_amdgcn_wmma_f32_16x16x32_f16(
                    false, af1, false, bfr, (short)0, acc[1][nt], false, false);
            }
        }
        // ---- K-step 1 ----
        {
            v16h af0 = SHUF16(a02, a03);
            v16h af1 = SHUF16(a12, a13);
#pragma unroll
            for (int nt = 0; nt < 4; ++nt) {
                const _Float16* bp = scur + (nt * 16 + bcol) * BPAD + 32 + kb;
                v8h b0 = *(const v8h*)(bp);
                v8h b1 = *(const v8h*)(bp + 16);
                v16h bfr = SHUF16(b0, b1);
                acc[0][nt] = __builtin_amdgcn_wmma_f32_16x16x32_f16(
                    false, af0, false, bfr, (short)0, acc[0][nt], false, false);
                acc[1][nt] = __builtin_amdgcn_wmma_f32_16x16x32_f16(
                    false, af1, false, bfr, (short)0, acc[1][nt], false, false);
            }
        }
        __syncthreads();   // readers done before buffer is overwritten
    }

    // D layout: VGPR r, lane -> row = 16*tile + (lane>>4)*8 + r, col = lane&15
#pragma unroll
    for (int mt = 0; mt < 2; ++mt) {
        const size_t mrow = (size_t)(mTile0 + mt) * 16 + (lane >> 4) * 8;
        const int    ncl  = ncol0 + r16;
#pragma unroll
        for (int nt = 0; nt < 4; ++nt) {
            _Float16* up = U + mrow * OUTW + ncl + nt * 16;
#pragma unroll
            for (int r = 0; r < 8; ++r)
                up[(size_t)r * OUTW] = (_Float16)(acc[mt][nt][r]);
        }
    }
}

// ---------------- 5. SRU scan + highway: U -> h_out (L,N,2H) f16 ----------------
__global__ __launch_bounds__(256)
void sru_scan(const _Float16* __restrict__ U, const float* __restrict__ bf,
              const float* __restrict__ br, _Float16* __restrict__ Hout) {
    int t   = blockIdx.x * 256 + threadIdx.x;   // 0 .. 2*N*H-1 = 131071
    int j   = t & (Hc - 1);
    int n   = (t >> 7) & (Nn - 1);
    int dir = t >> 16;
    float bfj = bf[dir * Hc + j];
    float brj = br[dir * Hc + j];
    const size_t cbase = (size_t)dir * FOURH + j;
    float c = 0.f;
    for (int s = 0; s < Lc; ++s) {
        int l = dir ? (Lc - 1 - s) : s;
        const _Float16* up = U + ((size_t)l * Nn + n) * OUTW + cbase;
        float z  = (float)up[0];
        float f  = (float)up[Hc];
        float r  = (float)up[2 * Hc];
        float hp = (float)up[3 * Hc];
        f = 1.f / (1.f + __expf(-(f + bfj)));
        r = 1.f / (1.f + __expf(-(r + brj)));
        c = f * c + (1.f - f) * z;
        float o = r * c + (1.f - r) * hp;
        Hout[((size_t)l * Nn + n) * (2 * Hc) + dir * Hc + j] = (_Float16)o;
    }
}

// ---------------- 6. ConvTranspose1d + bias + residual ----------------
// out(n,c,f) = sum_{j<2H, k} h3(f-k, n, j) * convW[j,c,k] + convb[c] + xn
// One block per n; h3[:, n, :] staged in LDS with padded row stride (260
// halves = 130 dwords) so lane reads at consecutive l hit distinct banks.
#define ROWP 260
__global__ __launch_bounds__(128)
void deconv_res(const _Float16* __restrict__ h3, const float* __restrict__ convW,
                const float* __restrict__ convb, const float* __restrict__ xn,
                float* __restrict__ out) {
    __shared__ _Float16 sh[Lc * ROWP];   // 62,920 B
    const int tid = threadIdx.x;         // = f
    const int n   = blockIdx.x;
    const int b   = n >> 7, t = n & (Tc - 1);

    for (int l = 0; l < Lc; ++l) {       // 256 halves per row, 128 dword copies
        const uint32_t* src = (const uint32_t*)(h3 + ((size_t)l * Nn + n) * 256);
        uint32_t* dst = (uint32_t*)(sh + l * ROWP);
        dst[tid] = src[tid];
    }
    __syncthreads();

    const int f    = tid;
    const int kmin = (f - (Lc - 1)) > 0 ? (f - (Lc - 1)) : 0;
    const int kmax = f < (Kk - 1) ? f : (Kk - 1);
    for (int c = 0; c < Cc; ++c) {
        float acc = 0.f;
        for (int j = 0; j < 2 * Hc; ++j) {
            const float* wp = convW + (size_t)j * (Cc * Kk) + c * Kk;  // uniform
            for (int k = kmin; k <= kmax; ++k)
                acc += (float)sh[(f - k) * ROWP + j] * wp[k];
        }
        size_t oidx = (((size_t)b * Cc + c) * Tc + t) * Fc + f;
        out[oidx] = acc + convb[c] + xn[oidx];
    }
}

// ---------------- launch ----------------
extern "C" void kernel_launch(void* const* d_in, const int* in_sizes, int n_in,
                              void* d_out, int out_size, void* d_ws, size_t ws_size,
                              hipStream_t stream) {
    const float* x     = (const float*)d_in[0];
    const float* gamma = (const float*)d_in[1];
    const float* beta  = (const float*)d_in[2];
    const float* W[4]  = { (const float*)d_in[3],  (const float*)d_in[6],
                           (const float*)d_in[9],  (const float*)d_in[12] };
    const float* bf[4] = { (const float*)d_in[4],  (const float*)d_in[7],
                           (const float*)d_in[10], (const float*)d_in[13] };
    const float* br[4] = { (const float*)d_in[5],  (const float*)d_in[8],
                           (const float*)d_in[11], (const float*)d_in[14] };
    const float* convW = (const float*)d_in[15];
    const float* convb = (const float*)d_in[16];
    float* out = (float*)d_out;

    // workspace carve-up (bytes, 256-aligned)
    char* ws = (char*)d_ws;
    float*    xn  = (float*)   (ws);                       // 16 MiB
    _Float16* hA  = (_Float16*)(ws + (size_t)16777216);    // M*512 f16 = 63.4 MB
    _Float16* hB  = (_Float16*)(ws + (size_t)80216064);    // M*256 f16 = 31.7 MB
    _Float16* U   = (_Float16*)(ws + (size_t)111935488);   // M*1024 f16 = 126.9 MB
    _Float16* Wt0 = (_Float16*)(ws + (size_t)238813184);   // 1024x512 f16
    _Float16* Wt1 = (_Float16*)(ws + (size_t)239861760);   // 1024x256 f16
    _Float16* Wt2 = (_Float16*)(ws + (size_t)240386048);
    _Float16* Wt3 = (_Float16*)(ws + (size_t)240910336);
    _Float16* Wt[4] = { Wt0, Wt1, Wt2, Wt3 };
    const int Kd[4] = { Cc * Kk, 2 * Hc, 2 * Hc, 2 * Hc };  // 512,256,256,256

    // 1. channel norm
    chan_norm<<<(Bc * Tc * Fc) / 256, 256, 0, stream>>>(x, gamma, beta, xn);

    // 2. unfold to h0 (into hA)
    unfold_k<<<(unsigned)((Mrows * 512) / 256), 256, 0, stream>>>(xn, hA);

    // 3. weight conversion
    for (int l = 0; l < 4; ++l) {
        int elems = OUTW * Kd[l];
        wconvert<<<(elems + 255) / 256, 256, 0, stream>>>(W[l], Wt[l], Kd[l]);
    }

    // 4. four SRU layers, ping-pong hA/hB
    _Float16* hin  = hA;
    _Float16* hout = hB;
    const dim3 gemmGrid((unsigned)(Mrows / 16 / 8), 8);    // 484 x 8
    for (int l = 0; l < 4; ++l) {
        sru_gemm<<<gemmGrid, 256, 0, stream>>>(hin, Wt[l], U, Kd[l]);
        sru_scan<<<(2 * Nn * Hc) / 256, 256, 0, stream>>>(U, bf[l], br[l], hout);
        _Float16* tmp = hin; hin = hout; hout = tmp;   // next in = this out
    }
    // after 4 swaps, final h3 lives in hA (L0:A->B, L1:B->A, L2:A->B, L3:B->A)

    // 5. transposed conv + bias + residual
    deconv_res<<<Nn, 128, 0, stream>>>(hA, convW, convb, xn, out);
}
